// Feedback_GNN_41326175322282
// MI455X (gfx1250) — matrix operand order repaired and checked by reading, not compiled, not run on hardware
//
#include <hip/hip_runtime.h>

#define BATCH   512
#define NUM_VN  2048
#define NUM_CN  1024
#define DEG     8
#define HID     32
#define MSG     16
#define NEDGE   (NUM_CN * DEG)
#define TILES   (BATCH / 16)        // 32 batch tiles of 16

typedef __attribute__((ext_vector_type(16))) _Float16 v16h;
typedef __attribute__((ext_vector_type(2)))  _Float16 v2h;
typedef __attribute__((ext_vector_type(8)))  float    v8f;

// ---- hardware tanh (gfx1250 V_TANH_F32 / V_TANH_F16), guarded --------------
#if __has_builtin(__builtin_amdgcn_tanhf)
#define TANH32(x) __builtin_amdgcn_tanhf(x)
#else
#define TANH32(x) tanhf(x)
#endif

#if __has_builtin(__builtin_amdgcn_tanhh)
#define TANH16(x) ((_Float16)__builtin_amdgcn_tanhh(x))
#elif __has_builtin(__builtin_amdgcn_tanh_f16)
#define TANH16(x) ((_Float16)__builtin_amdgcn_tanh_f16(x))
#else
#define TANH16(x) ((_Float16)TANH32((float)(x)))
#endif

// ---- WMMA fragment index helpers (CDNA5 16-bit layouts, wave32) -------------
// A 16x32 (MxK): lane m (0..15) = row m, halves K={0..7,16..23}; lanes 16..31 K+=8
__device__ __forceinline__ int a_kidx(int lane, int i) {
    int base = (i < 8) ? i : (i + 8);
    return base + ((lane >= 16) ? 8 : 0);
}
// B 32x16 (KxN): lane holds column n=lane&15; halves K={0..7,16..23}; lanes 16..31 K+=8
__device__ __forceinline__ int b_kidx(int lane, int i) {
    int base = (i < 8) ? i : (i + 8);
    return base + ((lane >= 16) ? 8 : 0);
}

__device__ __forceinline__ v8f zero8() {
    v8f z;
#pragma unroll
    for (int i = 0; i < 8; ++i) z[i] = 0.0f;
    return z;
}

__device__ __forceinline__ v8f wmma_f16(v16h a, v16h b, v8f c) {
    return __builtin_amdgcn_wmma_f32_16x16x32_f16(false, a, false, b, (short)0, c,
                                                  false, false);
}

// ---- CSR build + prep kernels ----------------------------------------------
__global__ void zero_kernel(int* ws, int n) {
    int t = blockIdx.x * blockDim.x + threadIdx.x;
    if (t < n) ws[t] = 0;
}

__global__ void slogit_kernel(const float* __restrict__ lx, const int* __restrict__ sx,
                              const float* __restrict__ lz, const int* __restrict__ sz,
                              float* __restrict__ ox, float* __restrict__ oz) {
    int t = blockIdx.x * blockDim.x + threadIdx.x;
    if (t < NUM_CN * BATCH) {
        ox[t] = lx[t] * (1.0f - 2.0f * (float)sx[t]);
        oz[t] = lz[t] * (1.0f - 2.0f * (float)sz[t]);
    }
}

__global__ void count_kernel(const int* __restrict__ evn_x,
                             const int* __restrict__ evn_z,
                             int* cnt_x, int* cnt_z) {
    int t = blockIdx.x * blockDim.x + threadIdx.x;
    if (t < NEDGE)          atomicAdd(&cnt_x[evn_x[t]], 1);
    else if (t < 2 * NEDGE) atomicAdd(&cnt_z[evn_z[t - NEDGE]], 1);
}

// One block (1024 threads) per branch; Hillis-Steele inclusive scan of 2048 counts.
__global__ void scan_kernel(const int* __restrict__ cnt_x, int* start_x,
                            const int* __restrict__ cnt_z, int* start_z) {
    const int* cnt = (blockIdx.x == 0) ? cnt_x : cnt_z;
    int* start     = (blockIdx.x == 0) ? start_x : start_z;
    __shared__ int sa[NUM_VN];
    __shared__ int sb[NUM_VN];
    int t = threadIdx.x;
    sa[t] = cnt[t];
    sa[t + 1024] = cnt[t + 1024];
    __syncthreads();
    int* src = sa;
    int* dst = sb;
    for (int off = 1; off < NUM_VN; off <<= 1) {
        for (int i = t; i < NUM_VN; i += 1024)
            dst[i] = src[i] + ((i >= off) ? src[i - off] : 0);
        __syncthreads();
        int* tmp = src; src = dst; dst = tmp;
    }
    if (t == 0) start[0] = 0;
    start[t + 1]        = src[t];
    start[t + 1 + 1024] = src[t + 1024];
}

__global__ void fill_kernel(const int* __restrict__ ecn_x, const int* __restrict__ evn_x,
                            const int* __restrict__ ecn_z, const int* __restrict__ evn_z,
                            const int* __restrict__ start_x, const int* __restrict__ start_z,
                            int* cur_x, int* cur_z, int* el_x, int* el_z) {
    int t = blockIdx.x * blockDim.x + threadIdx.x;
    if (t < NEDGE) {
        int vn = evn_x[t];
        int pos = atomicAdd(&cur_x[vn], 1);
        el_x[start_x[vn] + pos] = ecn_x[t];
    } else if (t < 2 * NEDGE) {
        int e = t - NEDGE;
        int vn = evn_z[e];
        int pos = atomicAdd(&cur_z[vn], 1);
        el_z[start_z[vn] + pos] = ecn_z[e];
    }
}

// ---- Fused kernel: per (vn, 16-batch tile) wave -----------------------------
__global__ __launch_bounds__(256)
void fused_gnn_kernel(const float* __restrict__ h_vn,
                      const float* __restrict__ slog_x, const float* __restrict__ slog_z,
                      const float* __restrict__ W1x, const float* __restrict__ W2x,
                      const float* __restrict__ W1z, const float* __restrict__ W2z,
                      const float* __restrict__ W_embed, const float* __restrict__ W_llr,
                      const int* __restrict__ start_x, const int* __restrict__ el_x,
                      const int* __restrict__ start_z, const int* __restrict__ el_z,
                      float* __restrict__ out) {
    __shared__ float lbuf[8][16 * 36];        // per-wave 16x36 staging tile

    const int lane  = threadIdx.x & 31;
    const int wslot = threadIdx.x >> 5;
    const int wid   = (blockIdx.x * blockDim.x + threadIdx.x) >> 5;
    const int vn    = wid >> 5;               // / TILES
    const int tile  = wid & (TILES - 1);
    const int b0    = tile * 16;
    const int m     = lane & 15;              // batch row within tile
    const int b     = b0 + m;
    const int col   = lane & 15;              // C-layout column held by this lane
    const int roff  = (lane >= 16) ? 8 : 0;   // C-layout row offset

    // h_vn features of THIS vn (source == destination vn in the reference)
    const float fv1 = h_vn[((size_t)b * NUM_VN + vn) * 3 + 0];
    const float fv2 = h_vn[((size_t)b * NUM_VN + vn) * 3 + 1];
    const float fv3 = h_vn[((size_t)b * NUM_VN + vn) * 3 + 2];

    // B fragments for W2x / W2z  ([HID=32 x MSG=16], K=32 exactly)
    v16h bW2x, bW2z;
#pragma unroll
    for (int i = 0; i < 16; ++i) {
        int k = b_kidx(lane, i);
        bW2x[i] = (_Float16)W2x[k * MSG + col];
        bW2z[i] = (_Float16)W2z[k * MSG + col];
    }

    auto run_branch = [&](const float* __restrict__ slog,
                          const float* __restrict__ W1, const v16h& bW2,
                          const int* __restrict__ start, const int* __restrict__ elist,
                          v8f& acc, float& invdeg) {
        // Prepack per-lane layer-1 weights: pair t covers hidden j0 = a_kidx(lane,2t), j0+1.
        // w1r0 = W1 row 0 (the h_cn feature); part = fv1..fv3 dot W1 rows 1..3 (edge-invariant).
        v2h w1r0[8], part[8];
#pragma unroll
        for (int t = 0; t < 8; ++t) {
            int j = a_kidx(lane, 2 * t);
            float p0 = fv1 * W1[HID + j]     + fv2 * W1[2 * HID + j]     + fv3 * W1[3 * HID + j];
            float p1 = fv1 * W1[HID + j + 1] + fv2 * W1[2 * HID + j + 1] + fv3 * W1[3 * HID + j + 1];
            part[t] = v2h{(_Float16)p0, (_Float16)p1};
            w1r0[t] = v2h{(_Float16)W1[j], (_Float16)W1[j + 1]};
        }
        int s0 = start[vn], s1 = start[vn + 1];
        invdeg = 1.0f / fmaxf((float)(s1 - s0), 1.0f);
        for (int p = s0; p < s1; ++p) {
            int cn = elist[p];
            _Float16 f0h = (_Float16)slog[cn * BATCH + b];
            v2h f0v = v2h{f0h, f0h};
            union { v16h v; v2h p2[8]; } A;
#pragma unroll
            for (int t = 0; t < 8; ++t) {
                v2h hs = f0v * w1r0[t] + part[t];   // v_pk_fma_f16
                A.p2[t] = v2h{TANH16(hs[0]), TANH16(hs[1])};
            }
            acc = wmma_f16(A.v, bW2, acc);          // C-chained segment sum
        }
    };

    v8f mx = zero8(), mz = zero8();
    float inv_x, inv_z;
    run_branch(slog_x, W1x, bW2x, start_x, el_x, mx, inv_x);
    run_branch(slog_z, W1z, bW2z, start_z, el_z, mz, inv_z);

    // Transpose C-layout messages into row-major feature tile [16 x 35] in LDS
    float* L = &lbuf[wslot][0];
#pragma unroll
    for (int r = 0; r < 8; ++r) {
        L[(roff + r) * 36 + col]      = mx[r] * inv_x;   // features 0..15  = m_x
        L[(roff + r) * 36 + 16 + col] = mz[r] * inv_z;   // features 16..31 = m_z
    }
    if (lane < 16) {                                     // features 32..34 = h_vn
        L[m * 36 + 32] = fv1; L[m * 36 + 33] = fv2;
        L[m * 36 + 34] = fv3; L[m * 36 + 35] = 0.0f;
    }
    __syncthreads();

    // GEMM1: [16x35(->64)] @ W_embed[35x32], two K-steps x two N-tiles
    v16h A0, A1;
#pragma unroll
    for (int i = 0; i < 16; ++i) {
        int k  = a_kidx(lane, i);
        A0[i] = (_Float16)L[m * 36 + k];
        int k2 = k + 32;
        A1[i] = (k2 < 35) ? (_Float16)L[m * 36 + k2] : (_Float16)0.0f;
    }
    v16h B00, B01, B10, B11;
#pragma unroll
    for (int i = 0; i < 16; ++i) {
        int k  = b_kidx(lane, i);
        B00[i] = (_Float16)W_embed[k * HID + col];
        B01[i] = (_Float16)W_embed[k * HID + 16 + col];
        int k2 = k + 32;
        B10[i] = (k2 < 35) ? (_Float16)W_embed[k2 * HID + col]      : (_Float16)0.0f;
        B11[i] = (k2 < 35) ? (_Float16)W_embed[k2 * HID + 16 + col] : (_Float16)0.0f;
    }
    v8f o0 = zero8(), o1 = zero8();
    o0 = wmma_f16(A0, B00, o0);
    o0 = wmma_f16(A1, B10, o0);
    o1 = wmma_f16(A0, B01, o1);
    o1 = wmma_f16(A1, B11, o1);

    // tanh (hardware v_tanh_f32) + transpose back through LDS for the LLR head
    __syncthreads();
#pragma unroll
    for (int r = 0; r < 8; ++r) {
        L[(roff + r) * 36 + col]      = TANH32(o0[r]);
        L[(roff + r) * 36 + 16 + col] = TANH32(o1[r]);
    }
    __syncthreads();

    // GEMM2: [16x32] @ W_llr[32x3 (padded to 16 cols)]
    v16h A2, B2;
#pragma unroll
    for (int i = 0; i < 16; ++i) {
        A2[i] = (_Float16)L[m * 36 + a_kidx(lane, i)];
        int k = b_kidx(lane, i);
        B2[i] = (col < 3) ? (_Float16)W_llr[k * 3 + col] : (_Float16)0.0f;
    }
    v8f o2 = wmma_f16(A2, B2, zero8());

    if (col < 3) {
#pragma unroll
        for (int r = 0; r < 8; ++r) {
            int brow = b0 + roff + r;
            out[((size_t)brow * NUM_VN + vn) * 3 + col] = o2[r];
        }
    }
}

// ---- launch -----------------------------------------------------------------
extern "C" void kernel_launch(void* const* d_in, const int* in_sizes, int n_in,
                              void* d_out, int out_size, void* d_ws, size_t ws_size,
                              hipStream_t stream) {
    const float* h_vn     = (const float*)d_in[0];
    const float* logit_hx = (const float*)d_in[1];
    const float* logit_hz = (const float*)d_in[2];
    const float* W1x      = (const float*)d_in[3];
    const float* W2x      = (const float*)d_in[4];
    const float* W1z      = (const float*)d_in[5];
    const float* W2z      = (const float*)d_in[6];
    const float* W_embed  = (const float*)d_in[7];
    const float* W_llr    = (const float*)d_in[8];
    const int*   synd_x   = (const int*)d_in[9];
    const int*   synd_z   = (const int*)d_in[10];
    const int*   ecn_x    = (const int*)d_in[11];
    const int*   evn_x    = (const int*)d_in[12];
    const int*   ecn_z    = (const int*)d_in[13];
    const int*   evn_z    = (const int*)d_in[14];

    int* ws      = (int*)d_ws;
    int* cnt_x   = ws;              // 2048
    int* cnt_z   = ws + 2048;       // 2048
    int* start_x = ws + 4096;       // 2049
    int* start_z = ws + 8192;       // 2049
    int* cur_x   = ws + 12288;      // 2048
    int* cur_z   = ws + 14336;      // 2048
    int* el_x    = ws + 16384;      // 8192
    int* el_z    = ws + 24576;      // 8192   (CSR total: 128 KB)
    float* slog_x = (float*)(ws + 32768);              // 1024*512 floats (2 MB)
    float* slog_z = slog_x + NUM_CN * BATCH;           // 2 MB

    zero_kernel<<<64, 256, 0, stream>>>(ws, 16384);
    slogit_kernel<<<(NUM_CN * BATCH) / 256, 256, 0, stream>>>(
        logit_hx, synd_x, logit_hz, synd_z, slog_x, slog_z);
    count_kernel<<<(2 * NEDGE) / 256, 256, 0, stream>>>(evn_x, evn_z, cnt_x, cnt_z);
    scan_kernel<<<2, 1024, 0, stream>>>(cnt_x, start_x, cnt_z, start_z);
    fill_kernel<<<(2 * NEDGE) / 256, 256, 0, stream>>>(ecn_x, evn_x, ecn_z, evn_z,
                                                       start_x, start_z,
                                                       cur_x, cur_z, el_x, el_z);
    // 2048 vn * 32 batch-tiles = 65536 waves; 8 waves (256 threads) per block
    fused_gnn_kernel<<<(NUM_VN * TILES) / 8, 256, 0, stream>>>(
        h_vn, slog_x, slog_z, W1x, W2x, W1z, W2z, W_embed, W_llr,
        start_x, el_x, start_z, el_z, (float*)d_out);
}
